// arGDPmax_net_34806414966964
// MI455X (gfx1250) — matrix operand order repaired and checked by reading, not compiled, not run on hardware
//
#include <hip/hip_runtime.h>
#include <hip/hip_bf16.h>
#include <math.h>

// ---------------- problem constants ----------------
constexpr int S_  = 128;     // scenarios
constexpr int WN_ = 3968;    // world agents
constexpr int N_  = 4096;    // S_ + WN_
constexpr int H_  = 512;     // hidden (4E)
constexpr int G_  = 1536;    // 3H gates
constexpr int DH_ = 512;     // decoder hidden
constexpr int T_  = 80;      // time steps
constexpr int NTG = G_ / 16; // 96 gate col tiles
constexpr int NTH = H_ / 16; // 32 hidden col tiles
constexpr int KT  = H_ / 32; // 16 k-tiles of 32

// ---------------- WMMA types ----------------
typedef __attribute__((ext_vector_type(16))) __bf16 v16bf;
typedef __attribute__((ext_vector_type(8)))  float  v8f;

union BF16x16 { v16bf v; uint4 q[2]; };

__device__ __forceinline__ unsigned short f2bf(float f) {
    union { float f; unsigned u; } x; x.f = f;
    unsigned r = x.u + 0x7FFFu + ((x.u >> 16) & 1u);   // RNE
    return (unsigned short)(r >> 16);
}

// A tile (16x32 bf16, row-major source, stride in elements).
// Lane layout: lanes 0-15 rows 0-15 K{0..7,16..23}; lanes 16-31 rows 0-15 K{8..15,24..31}
__device__ __forceinline__ v16bf load_a_tile(const unsigned short* A, int stride,
                                             int m0, int k0, int lane) {
    BF16x16 t;
    const unsigned short* p = A + (size_t)(m0 + (lane & 15)) * stride + k0 + ((lane >> 4) << 3);
    t.q[0] = *(const uint4*)(p);
    t.q[1] = *(const uint4*)(p + 16);
    return t.v;
}

// B tile from pre-packed weights: tile = 32 lanes x 16 bf16 contiguous per lane
__device__ __forceinline__ v16bf load_b_tile(const unsigned short* Bp, int tile, int lane) {
    BF16x16 t;
    const unsigned short* p = Bp + (size_t)tile * 512 + lane * 16;
    t.q[0] = *(const uint4*)(p);
    t.q[1] = *(const uint4*)(p + 8);
    return t.v;
}

#define WMMA_BF16(A, B, C) \
    __builtin_amdgcn_wmma_f32_16x16x32_bf16(false, (A), false, (B), (short)0, (C), false, false)

// ---------------- one-time kernels ----------------

// pack f32 weight [K, Ncols] -> bf16 WMMA-B layout tiles
__global__ void pack_b_kernel(const float* __restrict__ W, unsigned short* __restrict__ dst,
                              int K, int Ncols) {
    int i = blockIdx.x * blockDim.x + threadIdx.x;
    int total = K * Ncols;
    if (i >= total) return;
    int e    = i & 15;
    int lane = (i >> 4) & 31;
    int tile = i >> 9;
    int ntiles = Ncols >> 4;
    int kt = tile / ntiles, nt = tile - kt * ntiles;
    int krow = kt * 32 + ((lane >> 4) << 3) + (e & 7) + ((e >> 3) << 4);   // lanes16-31 hold K 8..15,24..31
    int ncol = nt * 16 + (lane & 15);
    dst[i] = f2bf(W[(size_t)krow * Ncols + ncol]);
}

__global__ void init_h_kernel(const float* __restrict__ sdc_h0, const float* __restrict__ wld_h0,
                              float* __restrict__ hf, unsigned short* __restrict__ hb) {
    int i = blockIdx.x * blockDim.x + threadIdx.x;
    if (i >= N_ * H_) return;
    float v = (i < S_ * H_) ? sdc_h0[i] : wld_h0[i - S_ * H_];
    hf[i] = v; hb[i] = f2bf(v);
}

// enc = concat([goals@gol_W+gol_b, dyn, ccls, int]) -> x buffer (bf16)
__global__ void encoder_kernel(const float* __restrict__ goals, const float* __restrict__ dyn,
                               const float* __restrict__ ccls, const float* __restrict__ itn,
                               const float* __restrict__ golW, const float* __restrict__ golb,
                               unsigned short* __restrict__ xbuf) {
    int i = blockIdx.x * blockDim.x + threadIdx.x;
    if (i >= N_ * H_) return;
    int n = i >> 9, c = i & 511;
    float v;
    if (c < 128)       v = goals[n * 2] * golW[c] + goals[n * 2 + 1] * golW[128 + c] + golb[c];
    else if (c < 256)  v = dyn [n * 128 + (c - 128)];
    else if (c < 384)  v = ccls[n * 128 + (c - 256)];
    else               v = itn [n * 128 + (c - 384)];
    xbuf[i] = f2bf(v);
}

// ---------------- per-step kernels ----------------

// segment_max (rows < S) / gather (rows >= S) -> x buffer (bf16)
__global__ void prep_kernel(const float* __restrict__ hf_prev, const int* __restrict__ seg,
                            unsigned short* __restrict__ xbuf) {
    int i = blockIdx.x * blockDim.x + threadIdx.x;
    if (i >= N_ * H_) return;
    int n = i >> 9, c = i & 511;
    float v;
    if (n < S_) {
        int lo = 0, hi = WN_;                     // lower_bound over sorted seg ids
        while (lo < hi) { int mid = (lo + hi) >> 1; if (seg[mid] < n) lo = mid + 1; else hi = mid; }
        float m = -INFINITY;
        for (int w = lo; w < WN_ && seg[w] == n; ++w)
            m = fmaxf(m, hf_prev[(size_t)(S_ + w) * H_ + c]);
        v = m;
    } else {
        int s = seg[n - S_];
        v = hf_prev[(size_t)s * H_ + c];
    }
    xbuf[i] = f2bf(v);
}

// One wave owns one 16x16 hidden tile: 6 accumulators (ir,iz,in,hr,hz,hn),
// K=512 reduction (16 WMMA steps x 6), fused GRU pointwise update.
// (6 accumulators is the sweet spot: 2x M-blocking was tried and made the
//  register allocator spill accumulators through v_mov_b64 + WMMA hazard NOPs.)
__global__ void __launch_bounds__(256)
gru_step_kernel(const unsigned short* __restrict__ xbuf,
                const unsigned short* __restrict__ hb_prev,
                const float* __restrict__ hf_prev,
                float* __restrict__ hf_next, unsigned short* __restrict__ hb_next,
                const unsigned short* __restrict__ sdcWi, const unsigned short* __restrict__ sdcWh,
                const unsigned short* __restrict__ wldWi, const unsigned short* __restrict__ wldWh,
                const float* __restrict__ sdc_bi, const float* __restrict__ sdc_bh,
                const float* __restrict__ wld_bi, const float* __restrict__ wld_bh) {
    int wid  = (blockIdx.x * blockDim.x + threadIdx.x) >> 5;
    int lane = threadIdx.x & 31;
    int mt = wid >> 5;          // 256 row tiles
    int nt = wid & 31;          // 32 hidden col tiles
    if (mt >= N_ / 16) return;
    bool sdc = (mt < S_ / 16);
    const unsigned short* Wi = sdc ? sdcWi : wldWi;
    const unsigned short* Wh = sdc ? sdcWh : wldWh;
    const float* bi = sdc ? sdc_bi : wld_bi;
    const float* bh = sdc ? sdc_bh : wld_bh;
    int m0 = mt * 16;

    v8f air = {}, aiz = {}, ain = {}, ahr = {}, ahz = {}, ahn = {};
    for (int kt = 0; kt < KT; ++kt) {
        if (kt + 1 < KT) {      // gfx1250 global_prefetch of next weight tiles
            __builtin_prefetch(Wi + (size_t)((kt + 1) * NTG + nt) * 512, 0, 1);
            __builtin_prefetch(Wh + (size_t)((kt + 1) * NTG + nt) * 512, 0, 1);
        }
        v16bf ax = load_a_tile(xbuf,    H_, m0, kt * 32, lane);
        v16bf ah = load_a_tile(hb_prev, H_, m0, kt * 32, lane);
        v16bf b;
        b = load_b_tile(Wi, kt * NTG + nt,      lane); air = WMMA_BF16(ax, b, air);
        b = load_b_tile(Wi, kt * NTG + nt + 32, lane); aiz = WMMA_BF16(ax, b, aiz);
        b = load_b_tile(Wi, kt * NTG + nt + 64, lane); ain = WMMA_BF16(ax, b, ain);
        b = load_b_tile(Wh, kt * NTG + nt,      lane); ahr = WMMA_BF16(ah, b, ahr);
        b = load_b_tile(Wh, kt * NTG + nt + 32, lane); ahz = WMMA_BF16(ah, b, ahz);
        b = load_b_tile(Wh, kt * NTG + nt + 64, lane); ahn = WMMA_BF16(ah, b, ahn);
    }
    int col = nt * 16 + (lane & 15);
    float bir = bi[col], biz = bi[col + 512], bin_ = bi[col + 1024];
    float bhr = bh[col], bhz = bh[col + 512], bhn  = bh[col + 1024];
    int rbase = m0 + ((lane >> 4) << 3);            // C layout: VGPR d -> row d (+8 for hi lanes)
    #pragma unroll
    for (int d = 0; d < 8; ++d) {
        int row = rbase + d;
        float r  = 1.f / (1.f + __expf(-((air[d] + bir) + (ahr[d] + bhr))));
        float z  = 1.f / (1.f + __expf(-((aiz[d] + biz) + (ahz[d] + bhz))));
        float nn = tanhf((ain[d] + bin_) + r * (ahn[d] + bhn));
        float hp = hf_prev[(size_t)row * H_ + col];
        float hn2 = (1.f - z) * nn + z * hp;
        hf_next[(size_t)row * H_ + col] = hn2;
        hb_next[(size_t)row * H_ + col] = f2bf(hn2);
    }
}

// hidden = leaky_relu(h @ o1W + o1b, 0.1)
__global__ void __launch_bounds__(256)
mlp_hidden_kernel(const unsigned short* __restrict__ hb,
                  const unsigned short* __restrict__ sdcW, const unsigned short* __restrict__ wldW,
                  const float* __restrict__ sdc_b, const float* __restrict__ wld_b,
                  float* __restrict__ hidden) {
    int wid  = (blockIdx.x * blockDim.x + threadIdx.x) >> 5;
    int lane = threadIdx.x & 31;
    int mt = wid >> 5, nt = wid & 31;
    if (mt >= N_ / 16) return;
    bool sdc = (mt < S_ / 16);
    const unsigned short* Wp = sdc ? sdcW : wldW;
    const float* b1 = sdc ? sdc_b : wld_b;
    int m0 = mt * 16;
    v8f acc = {};
    for (int kt = 0; kt < KT; ++kt) {
        if (kt + 1 < KT)
            __builtin_prefetch(Wp + (size_t)((kt + 1) * NTH + nt) * 512, 0, 1);
        v16bf a = load_a_tile(hb, H_, m0, kt * 32, lane);
        v16bf b = load_b_tile(Wp, kt * NTH + nt, lane);
        acc = WMMA_BF16(a, b, acc);
    }
    int col = nt * 16 + (lane & 15);
    float bb = b1[col];
    int rbase = m0 + ((lane >> 4) << 3);
    #pragma unroll
    for (int d = 0; d < 8; ++d) {
        float v = acc[d] + bb;
        v = (v > 0.f) ? v : 0.1f * v;
        hidden[(size_t)(rbase + d) * DH_ + col] = v;
    }
}

// out[:, t, :] = hidden @ o2W + o2b   (OD=2 -> plain dot per (row, od))
__global__ void mlp_out_kernel(const float* __restrict__ hidden,
                               const float* __restrict__ sdcW, const float* __restrict__ sdc_b,
                               const float* __restrict__ wldW, const float* __restrict__ wld_b,
                               float* __restrict__ out, int t) {
    int i = blockIdx.x * blockDim.x + threadIdx.x;
    if (i >= N_ * 2) return;
    int n = i >> 1, od = i & 1;
    const float* W2 = (n < S_) ? sdcW : wldW;
    float acc = (n < S_) ? sdc_b[od] : wld_b[od];
    const float* hr = hidden + (size_t)n * DH_;
    #pragma unroll 8
    for (int c = 0; c < DH_; ++c) acc += hr[c] * W2[c * 2 + od];
    out[(size_t)n * (T_ * 2) + t * 2 + od] = acc;
}

// ---------------- host ----------------
extern "C" void kernel_launch(void* const* d_in, const int* in_sizes, int n_in,
                              void* d_out, int out_size, void* d_ws, size_t ws_size,
                              hipStream_t stream) {
    const float* goals   = (const float*)d_in[0];
    const float* agn_dyn = (const float*)d_in[1];
    const float* agn_ccl = (const float*)d_in[2];
    const float* agn_int = (const float*)d_in[3];
    const int*   seg_ids = (const int*)  d_in[4];
    const float* sdc_h0  = (const float*)d_in[5];
    const float* wld_h0  = (const float*)d_in[6];
    const float* gol_W   = (const float*)d_in[7];
    const float* gol_b   = (const float*)d_in[8];
    const float* sdc_Wi  = (const float*)d_in[9];
    const float* sdc_Wh  = (const float*)d_in[10];
    const float* sdc_bi  = (const float*)d_in[11];
    const float* sdc_bh  = (const float*)d_in[12];
    const float* wld_Wi  = (const float*)d_in[13];
    const float* wld_Wh  = (const float*)d_in[14];
    const float* wld_bi  = (const float*)d_in[15];
    const float* wld_bh  = (const float*)d_in[16];
    const float* sdc_o1W = (const float*)d_in[17];
    const float* sdc_o1b = (const float*)d_in[18];
    const float* sdc_o2W = (const float*)d_in[19];
    const float* sdc_o2b = (const float*)d_in[20];
    const float* wld_o1W = (const float*)d_in[21];
    const float* wld_o1b = (const float*)d_in[22];
    const float* wld_o2W = (const float*)d_in[23];
    const float* wld_o2b = (const float*)d_in[24];
    float* out = (float*)d_out;

    // workspace layout (bytes)
    char* ws = (char*)d_ws;
    size_t off = 0;
    auto alloc = [&](size_t bytes) { char* p = ws + off; off = (off + bytes + 255) & ~(size_t)255; return p; };
    unsigned short* sdcWi_p  = (unsigned short*)alloc((size_t)H_ * G_ * 2);
    unsigned short* sdcWh_p  = (unsigned short*)alloc((size_t)H_ * G_ * 2);
    unsigned short* wldWi_p  = (unsigned short*)alloc((size_t)H_ * G_ * 2);
    unsigned short* wldWh_p  = (unsigned short*)alloc((size_t)H_ * G_ * 2);
    unsigned short* sdcO1_p  = (unsigned short*)alloc((size_t)H_ * DH_ * 2);
    unsigned short* wldO1_p  = (unsigned short*)alloc((size_t)H_ * DH_ * 2);
    float*          hf[2];
    hf[0] = (float*)alloc((size_t)N_ * H_ * 4);
    hf[1] = (float*)alloc((size_t)N_ * H_ * 4);
    unsigned short* hb[2];
    hb[0] = (unsigned short*)alloc((size_t)N_ * H_ * 2);
    hb[1] = (unsigned short*)alloc((size_t)N_ * H_ * 2);
    unsigned short* xbuf   = (unsigned short*)alloc((size_t)N_ * H_ * 2);
    float*          hidden = (float*)alloc((size_t)N_ * DH_ * 4);
    (void)ws_size;

    const int TB = 256;
    // one-time: pack weights to bf16 WMMA-B layout
    {
        int tot = H_ * G_, blk = (tot + TB - 1) / TB;
        pack_b_kernel<<<blk, TB, 0, stream>>>(sdc_Wi, sdcWi_p, H_, G_);
        pack_b_kernel<<<blk, TB, 0, stream>>>(sdc_Wh, sdcWh_p, H_, G_);
        pack_b_kernel<<<blk, TB, 0, stream>>>(wld_Wi, wldWi_p, H_, G_);
        pack_b_kernel<<<blk, TB, 0, stream>>>(wld_Wh, wldWh_p, H_, G_);
        int tot2 = H_ * DH_, blk2 = (tot2 + TB - 1) / TB;
        pack_b_kernel<<<blk2, TB, 0, stream>>>(sdc_o1W, sdcO1_p, H_, DH_);
        pack_b_kernel<<<blk2, TB, 0, stream>>>(wld_o1W, wldO1_p, H_, DH_);
    }
    int nhBlocks = (N_ * H_ + TB - 1) / TB;
    init_h_kernel<<<nhBlocks, TB, 0, stream>>>(sdc_h0, wld_h0, hf[0], hb[0]);
    encoder_kernel<<<nhBlocks, TB, 0, stream>>>(goals, agn_dyn, agn_ccl, agn_int,
                                                gol_W, gol_b, xbuf);

    int gemmBlocks = (N_ / 16) * NTH / 8;   // 8 waves/block, one wave per 16x16 tile
    for (int t = 0; t < T_; ++t) {
        int cur = t & 1, nxt = (t + 1) & 1;
        if (t > 0)
            prep_kernel<<<nhBlocks, TB, 0, stream>>>(hf[cur], seg_ids, xbuf);
        gru_step_kernel<<<gemmBlocks, TB, 0, stream>>>(
            xbuf, hb[cur], hf[cur], hf[nxt], hb[nxt],
            sdcWi_p, sdcWh_p, wldWi_p, wldWh_p,
            sdc_bi, sdc_bh, wld_bi, wld_bh);
        mlp_hidden_kernel<<<gemmBlocks, TB, 0, stream>>>(
            hb[nxt], sdcO1_p, wldO1_p, sdc_o1b, wld_o1b, hidden);
        mlp_out_kernel<<<(N_ * 2 + TB - 1) / TB, TB, 0, stream>>>(
            hidden, sdc_o2W, sdc_o2b, wld_o2W, wld_o2b, out, t);
    }
}
